// SoftEmbedMap_56882546868728
// MI455X (gfx1250) — compile-verified, matrix operand order checked by reading, not run on hardware
//
#include <hip/hip_runtime.h>

#define ASR_DIM 1024
#define VOCAB   32000
#define LLM_DIM 4096
#define NROWS   8192   // 4 * 2048

typedef __attribute__((ext_vector_type(16))) __bf16 v16bf;
typedef __attribute__((ext_vector_type(8)))  float  v8f;

union FragU { v16bf v; uint4 q[2]; };

// float -> bf16, round to nearest even
__device__ __forceinline__ unsigned short f2bf(float x) {
    unsigned int u = __float_as_uint(x);
    u += 0x7FFFu + ((u >> 16) & 1u);
    return (unsigned short)(u >> 16);
}

// Load a 16x32 bf16 WMMA fragment (A-layout, or B-layout for K-contiguous data).
// Caller passes p = base + row*ld + k0 + (lane>>4)*8.
// ISA layout: regs v0..v3 hold k = h*8 .. h*8+7 (packed pairs), v4..v7 hold k+16.
__device__ __forceinline__ v16bf frag_ld(const unsigned short* p) {
    FragU f;
    f.q[0] = *(const uint4*)(p);
    f.q[1] = *(const uint4*)(p + 16);
    return f.v;
}

__device__ __forceinline__ v8f wmma_bf16(v16bf a, v16bf b, v8f c) {
    return __builtin_amdgcn_wmma_f32_16x16x32_bf16(false, a, false, b, (short)0, c,
                                                   false, false);
}

// ---------------------------------------------------------------- convert f32 -> bf16
__global__ void k_cvt4(const float* __restrict__ in, unsigned short* __restrict__ out,
                       long n4) {
    long i = (long)blockIdx.x * blockDim.x + threadIdx.x;
    if (i >= n4) return;
    float4 f = ((const float4*)in)[i];
    unsigned int lo = (unsigned int)f2bf(f.x) | ((unsigned int)f2bf(f.y) << 16);
    unsigned int hi = (unsigned int)f2bf(f.z) | ((unsigned int)f2bf(f.w) << 16);
    ((uint2*)out)[i] = make_uint2(lo, hi);
}

// ------------------------------------------------- transpose-convert embed_w -> E^T bf16
// in:  [VOCAB][LLM_DIM] f32   out: [LLM_DIM][VOCAB] bf16
__global__ void k_transpose_cvt(const float* __restrict__ in,
                                unsigned short* __restrict__ out) {
    __shared__ float tile[32][33];
    const int v0 = blockIdx.x * 32;
    const int e0 = blockIdx.y * 32;
    const int tx = threadIdx.x, ty = threadIdx.y;   // block 32x8
#pragma unroll
    for (int r = 0; r < 4; ++r) {
        int v = ty + r * 8;
        tile[v][tx] = in[(long)(v0 + v) * LLM_DIM + e0 + tx];
    }
    __syncthreads();
#pragma unroll
    for (int r = 0; r < 4; ++r) {
        int e = ty + r * 8;
        out[(long)(e0 + e) * VOCAB + v0 + tx] = f2bf(tile[tx][e]);
    }
}

// ------------------------------------------------------------------- softmax statistics
// One workgroup per 16-row tile; 8 waves split the vocab; online (max, sum-exp).
__global__ __launch_bounds__(256) void k_stats(const unsigned short* __restrict__ A,
                                               const unsigned short* __restrict__ W,
                                               const float* __restrict__ bias,
                                               float* __restrict__ rowmax,
                                               float* __restrict__ rowsum) {
    __shared__ unsigned short sA[16 * ASR_DIM];   // 32 KB
    __shared__ float smM[8][16], smS[8][16];
    const int mbase = blockIdx.x * 16;
    const int tid = threadIdx.x, lane = tid & 31, wave = tid >> 5;
    const int n = lane & 15, h = lane >> 4;

    {   // stage A tile into LDS (16 x 1024 bf16)
        const uint4* src = (const uint4*)(A + (long)mbase * ASR_DIM);
        uint4* dst = (uint4*)sA;
        for (int i = tid; i < 16 * ASR_DIM / 8; i += 256) dst[i] = src[i];
    }
    __syncthreads();

    float rm[8], rs[8];
#pragma unroll
    for (int j = 0; j < 8; ++j) { rm[j] = -INFINITY; rs[j] = 0.f; }

    for (int c = wave; c < VOCAB / 16; c += 8) {
        const int v0 = c * 16;
        const unsigned short* wrow = W + (long)(v0 + n) * ASR_DIM + h * 8;
        if (v0 + 128 < VOCAB)   // prefetch next chunk's row for this lane
            __builtin_prefetch(W + (long)(v0 + 128 + n) * ASR_DIM, 0, 1);
        v8f acc = {};
#pragma unroll 8
        for (int ks = 0; ks < ASR_DIM / 32; ++ks) {
            v16bf af = frag_ld(sA + n * ASR_DIM + ks * 32 + h * 8);
            v16bf bf_ = frag_ld(wrow + ks * 32);
            acc = wmma_bf16(af, bf_, acc);
        }
        const float bz = bias[v0 + n];
#pragma unroll
        for (int j = 0; j < 8; ++j) {
            float x = acc[j] + bz;
            float cm = x;
#pragma unroll
            for (int m = 1; m < 16; m <<= 1) cm = fmaxf(cm, __shfl_xor(cm, m, 32));
            float nm = fmaxf(rm[j], cm);
            float e = __expf(x - nm);
#pragma unroll
            for (int m = 1; m < 16; m <<= 1) e += __shfl_xor(e, m, 32);
            rs[j] = rs[j] * __expf(rm[j] - nm) + e;
            rm[j] = nm;
        }
    }
    if (n == 0) {
#pragma unroll
        for (int j = 0; j < 8; ++j) {
            smM[wave][j + 8 * h] = rm[j];
            smS[wave][j + 8 * h] = rs[j];
        }
    }
    __syncthreads();
    if (tid < 16) {
        float M = -INFINITY;
        for (int w = 0; w < 8; ++w) M = fmaxf(M, smM[w][tid]);
        float S = 0.f;
        for (int w = 0; w < 8; ++w) S += smS[w][tid] * __expf(smM[w][tid] - M);
        rowmax[mbase + tid] = M;
        rowsum[mbase + tid] = S;
    }
}

// ------------------------------------------------------------------ fused probs @ embed
// Workgroup: 16 waves, 16 rows, full 4096 output columns in registers (16 v8f / lane).
__global__ __launch_bounds__(512) void k_map(const unsigned short* __restrict__ A,
                                             const unsigned short* __restrict__ W,
                                             const unsigned short* __restrict__ ET,
                                             const float* __restrict__ bias,
                                             const float* __restrict__ rowmax,
                                             const float* __restrict__ rowsum,
                                             float* __restrict__ out) {
    __shared__ unsigned short sA[16 * ASR_DIM];   // 32 KB
    __shared__ unsigned short sP[16 * 512];       // 16 KB probs tile (bf16)
    const int mbase = blockIdx.x * 16;
    const int tid = threadIdx.x, lane = tid & 31, wave = tid >> 5;   // 16 waves
    const int n = lane & 15, h = lane >> 4;

    {
        const uint4* src = (const uint4*)(A + (long)mbase * ASR_DIM);
        uint4* dst = (uint4*)sA;
        for (int i = tid; i < 16 * ASR_DIM / 8; i += 512) dst[i] = src[i];
    }
    float rmax8[8], rinv8[8];
#pragma unroll
    for (int j = 0; j < 8; ++j) {
        int r = mbase + j + 8 * h;
        rmax8[j] = rowmax[r];
        rinv8[j] = 1.f / rowsum[r];
    }
    v8f acc[16];
#pragma unroll
    for (int t = 0; t < 16; ++t) acc[t] = (v8f){};
    __syncthreads();

    const int NIT = (VOCAB + 511) / 512;   // 63 (last block partially valid)
    for (int it = 0; it < NIT; ++it) {
        const int v0 = it * 512 + wave * 32;
        // ---- stage 1: this wave's 32 logit columns -> probs into LDS ----
        const int c0 = v0 + n, c1 = v0 + 16 + n;
        const int c0c = c0 < VOCAB ? c0 : VOCAB - 1;
        const int c1c = c1 < VOCAB ? c1 : VOCAB - 1;
        const unsigned short* wrow0 = W + (long)c0c * ASR_DIM + h * 8;
        const unsigned short* wrow1 = W + (long)c1c * ASR_DIM + h * 8;
        v8f l0 = {}, l1 = {};
#pragma unroll 8
        for (int ks = 0; ks < ASR_DIM / 32; ++ks) {
            v16bf af = frag_ld(sA + n * ASR_DIM + ks * 32 + h * 8);
            l0 = wmma_bf16(af, frag_ld(wrow0 + ks * 32), l0);
            l1 = wmma_bf16(af, frag_ld(wrow1 + ks * 32), l1);
        }
        const float b0 = bias[c0c], b1 = bias[c1c];
#pragma unroll
        for (int j = 0; j < 8; ++j) {
            float x0 = l0[j] + b0;
            float x1 = l1[j] + b1;
            float p0 = (c0 < VOCAB) ? __expf(x0 - rmax8[j]) * rinv8[j] : 0.f;
            float p1 = (c1 < VOCAB) ? __expf(x1 - rmax8[j]) * rinv8[j] : 0.f;
            const int row = j + 8 * h;
            sP[row * 512 + wave * 32 + n] = f2bf(p0);
            sP[row * 512 + wave * 32 + 16 + n] = f2bf(p1);
        }
        __syncthreads();
        // ---- stage 2: acc += probs(16x512) @ E^T slice (512 x 256 for this wave) ----
#pragma unroll 1
        for (int cc = 0; cc < 16; ++cc) {
            v16bf pf = frag_ld(sP + n * 512 + cc * 32 + h * 8);
            const long vbase = (long)it * 512 + cc * 32;
#pragma unroll
            for (int nt = 0; nt < 16; ++nt) {
                const int col = wave * 256 + nt * 16 + n;   // output-dim index
                v16bf ef = frag_ld(ET + (long)col * VOCAB + vbase + h * 8);
                acc[nt] = wmma_bf16(pf, ef, acc[nt]);
            }
        }
        __syncthreads();
    }
    // ---- write out [8192][4096] f32 ----
#pragma unroll
    for (int nt = 0; nt < 16; ++nt) {
        const int col = wave * 256 + nt * 16 + n;
#pragma unroll
        for (int j = 0; j < 8; ++j)
            out[(long)(mbase + j + 8 * h) * LLM_DIM + col] = acc[nt][j];
    }
}

// -------------------------------------------------------------------------- launcher
extern "C" void kernel_launch(void* const* d_in, const int* in_sizes, int n_in,
                              void* d_out, int out_size, void* d_ws, size_t ws_size,
                              hipStream_t stream) {
    const float* asr = (const float*)d_in[0];   // [4,2048,1024]
    const float* pw  = (const float*)d_in[1];   // [32000,1024]
    const float* pb  = (const float*)d_in[2];   // [32000]
    const float* ew  = (const float*)d_in[3];   // [32000,4096]
    float* out = (float*)d_out;                 // [4,2048,4096]

    char* ws = (char*)d_ws;
    const size_t off_a  = 0;                                        // 16 MiB
    const size_t off_w  = off_a + (size_t)NROWS * ASR_DIM * 2;      // 62.5 MiB
    const size_t off_eT = off_w + (size_t)VOCAB * ASR_DIM * 2;      // 250 MiB
    const size_t off_rm = off_eT + (size_t)LLM_DIM * VOCAB * 2 + 4096;  // pad for tail frags
    const size_t off_rs = off_rm + (size_t)NROWS * 4;

    unsigned short* a_bf = (unsigned short*)(ws + off_a);
    unsigned short* w_bf = (unsigned short*)(ws + off_w);
    unsigned short* eT   = (unsigned short*)(ws + off_eT);
    float* rmax = (float*)(ws + off_rm);
    float* rsum = (float*)(ws + off_rs);

    {   // A f32 -> bf16
        long n4 = (long)NROWS * ASR_DIM / 4;
        k_cvt4<<<(unsigned)((n4 + 255) / 256), 256, 0, stream>>>(asr, a_bf, n4);
    }
    {   // W f32 -> bf16
        long n4 = (long)VOCAB * ASR_DIM / 4;
        k_cvt4<<<(unsigned)((n4 + 255) / 256), 256, 0, stream>>>(pw, w_bf, n4);
    }
    // E -> E^T bf16
    k_transpose_cvt<<<dim3(VOCAB / 32, LLM_DIM / 32), dim3(32, 8), 0, stream>>>(ew, eT);
    // softmax stats
    k_stats<<<NROWS / 16, 256, 0, stream>>>(a_bf, w_bf, pb, rmax, rsum);
    // fused probs @ embed
    k_map<<<NROWS / 16, 512, 0, stream>>>(a_bf, w_bf, eT, pb, rmax, rsum, out);
}
